// GRU_5798205849835
// MI455X (gfx1250) — compile-verified, matrix operand order checked by reading, not run on hardware
//
#include <hip/hip_runtime.h>
#include <hip/hip_bf16.h>
#include <math.h>

// ---------------------------------------------------------------------------
// GRU (2 layers) + FC for MI455X / gfx1250, wave32 WMMA.
//   B=8192, T=60, D=158, H=32, G=3H=96
// Kernel 1: xg0 = x @ w_ih_l0^T + b_ih_l0   (fp32 WMMA 16x16x4, K padded 160)
// Kernel 2: fused 2-layer GRU scan          (bf16 WMMA 16x16x32, fp32 accum)
// Kernel 3: out = h2[:, -1, :] @ fc_w^T + fc_b
// ---------------------------------------------------------------------------

typedef __attribute__((ext_vector_type(2)))  float   v2f;
typedef __attribute__((ext_vector_type(8)))  float   v8f;
typedef __attribute__((ext_vector_type(16))) __bf16  v16bf;

namespace gru {
constexpr int B = 8192;
constexpr int T = 60;
constexpr int D = 158;
constexpr int H = 32;
constexpr int G = 96;          // 3*H, gate order [r|z|n]
constexpr int MT = 16;         // batch rows per wave (WMMA M)
}

__device__ __forceinline__ v8f splat8(float x) {
    v8f r;
#pragma unroll
    for (int j = 0; j < 8; ++j) r[j] = x;
    return r;
}

__device__ __forceinline__ float sigmoidf_(float x) {
    return 1.0f / (1.0f + __expf(-x));
}

// ---------------------------------------------------------------------------
// Kernel 1: layer-0 input-gate GEMM.
//   M = B*T (tile = btile*T + t), N = 96, K = 158 (padded to 160).
//   A fragment (f32 16x4): lanes 0-15 rows M, VGPR0/1 = K {k,k+1}; lanes 16-31
//   same rows, K {k+2,k+3}.  B fragment mirrors with N across lanes.
//   Main loop k=0..152 needs no bounds checks (max index 155 < 158); the final
//   K-step zeroes the upper lane-half branchlessly (v_cndmask, no EXEC forks).
//   Output layout: xg0[tile][row 0..15][g 0..95]   (tile = btile*T + t)
// ---------------------------------------------------------------------------
__global__ __launch_bounds__(256) void gru_xgemm0(
    const float* __restrict__ x,       // [B][T][D]
    const float* __restrict__ w_ih,    // [G][D]
    const float* __restrict__ b_ih,    // [G]
    float* __restrict__ xg0)           // [B/16 * T][16][G]
{
    const int lane   = threadIdx.x & 31;
    const int wave   = threadIdx.x >> 5;
    const int tile   = blockIdx.x * 8 + wave;          // 0 .. 30719
    const int btile  = tile / gru::T;
    const int t      = tile - btile * gru::T;
    const int lanelo = lane & 15;
    const int hi     = lane >> 4;

    // Per-lane base pointers (8-byte aligned: row starts and ka are even).
    const float* arow =
        x + ((size_t)(btile * gru::MT + lanelo) * gru::T + t) * gru::D + 2 * hi;
    const float* brow = w_ih + (size_t)lanelo * gru::D + 2 * hi;

    v8f acc[6];
#pragma unroll
    for (int nt = 0; nt < 6; ++nt) acc[nt] = splat8(0.0f);

    // 39 unguarded K-steps: b64 loads, immediate offsets for the 6 N-tiles.
    for (int k = 0; k < 156; k += 4) {
        const v2f a = *(const v2f*)(arow + k);
#pragma unroll
        for (int nt = 0; nt < 6; ++nt) {
            const v2f b = *(const v2f*)(brow + (size_t)nt * 16 * gru::D + k);
            acc[nt] = __builtin_amdgcn_wmma_f32_16x16x4_f32(
                false, a, false, b, (short)0, acc[nt], false, false);
        }
    }

    // Epilogue k=156: lanes 16-31 would read K=158,159 -> substitute zeros.
    {
        const v2f al = *(const v2f*)(arow - 2 * hi + 156);   // K = {156,157}
        v2f a;
        a[0] = hi ? 0.0f : al[0];
        a[1] = hi ? 0.0f : al[1];
#pragma unroll
        for (int nt = 0; nt < 6; ++nt) {
            const v2f bl =
                *(const v2f*)(brow - 2 * hi + (size_t)nt * 16 * gru::D + 156);
            v2f b;
            b[0] = hi ? 0.0f : bl[0];
            b[1] = hi ? 0.0f : bl[1];
            acc[nt] = __builtin_amdgcn_wmma_f32_16x16x4_f32(
                false, a, false, b, (short)0, acc[nt], false, false);
        }
    }

    float* outp = xg0 + (size_t)tile * gru::MT * gru::G;
#pragma unroll
    for (int nt = 0; nt < 6; ++nt) {
        const float bv = b_ih[nt * 16 + lanelo];
#pragma unroll
        for (int j = 0; j < 8; ++j)                    // C: row j / j+8 per half
            outp[(j + 8 * hi) * gru::G + nt * 16 + lanelo] = acc[nt][j] + bv;
    }
}

// ---------------------------------------------------------------------------
// Kernel 2 helpers: bf16 fragment packing per ISA 16-bit layouts:
//   K pattern per VGPR v, lane-half hi:  k0 = 2*(v&3) + 16*(v>>2) + 8*hi
// ---------------------------------------------------------------------------
__device__ __forceinline__ v16bf load_w_frag(
    const float* __restrict__ W,   // [G][H] row-major (PyTorch w.T on the fly)
    int nt, int lanelo, int hi)
{
    v16bf f;
    const int n = nt * 16 + lanelo;                    // output column
#pragma unroll
    for (int v = 0; v < 8; ++v) {
        const int k0 = 2 * (v & 3) + 16 * (v >> 2) + 8 * hi;
        f[2 * v]     = (__bf16)W[(size_t)n * gru::H + k0];
        f[2 * v + 1] = (__bf16)W[(size_t)n * gru::H + k0 + 1];
    }
    return f;
}

// Re-pack fp32 hidden state (C-layout: h[c] reg j = row j+8*hi, col c*16+lanelo)
// into the 16-bit A-fragment layout via a 2 KB LDS transpose round-trip.
__device__ __forceinline__ v16bf build_a_frag(
    const v8f h[2], float* __restrict__ sA, int lanelo, int hi)
{
    __syncthreads();
#pragma unroll
    for (int c = 0; c < 2; ++c)
#pragma unroll
        for (int j = 0; j < 8; ++j)
            sA[(j + 8 * hi) * gru::H + c * 16 + lanelo] = h[c][j];
    __syncthreads();
    v16bf a;
#pragma unroll
    for (int v = 0; v < 8; ++v) {
        const int k0 = 2 * (v & 3) + 16 * (v >> 2) + 8 * hi;
        a[2 * v]     = (__bf16)sA[lanelo * gru::H + k0];
        a[2 * v + 1] = (__bf16)sA[lanelo * gru::H + k0 + 1];
    }
    return a;
}

// ---------------------------------------------------------------------------
// Kernel 2: fused 2-layer GRU scan. One wave = 16 batch rows, 60 time steps.
// All 18 recurrent-side weight fragments live in VGPRs for the whole kernel.
// ---------------------------------------------------------------------------
__global__ __launch_bounds__(32) void gru_scan(
    const float* __restrict__ xg0,     // [B/16 * T][16][G]  (bias pre-folded)
    const float* __restrict__ w_hh0,   // [G][H]
    const float* __restrict__ b_hh0,   // [G]
    const float* __restrict__ w_ih1,   // [G][H]
    const float* __restrict__ b_ih1,   // [G]
    const float* __restrict__ w_hh1,   // [G][H]
    const float* __restrict__ b_hh1,   // [G]
    float* __restrict__ h2_last)       // [B][H]
{
    __shared__ float sA[gru::MT * gru::H];             // 2 KB transpose buffer

    const int lane   = threadIdx.x;
    const int lanelo = lane & 15;
    const int hi     = lane >> 4;
    const int btile  = blockIdx.x;

    // Hoisted bf16 B-fragments (3 matrices x 6 N-tiles x 8 VGPRs = 144 VGPRs).
    v16bf Whh0[6], Wih1[6], Whh1[6];
    float bh0[6], bi1[6], bh1[6];
#pragma unroll
    for (int nt = 0; nt < 6; ++nt) {
        Whh0[nt] = load_w_frag(w_hh0, nt, lanelo, hi);
        Wih1[nt] = load_w_frag(w_ih1, nt, lanelo, hi);
        Whh1[nt] = load_w_frag(w_hh1, nt, lanelo, hi);
        bh0[nt]  = b_hh0[nt * 16 + lanelo];
        bi1[nt]  = b_ih1[nt * 16 + lanelo];
        bh1[nt]  = b_hh1[nt * 16 + lanelo];
    }

    v8f h1[2], h2[2];
#pragma unroll
    for (int c = 0; c < 2; ++c) { h1[c] = splat8(0.0f); h2[c] = splat8(0.0f); }
    v16bf a1, a2;                                      // bf16 A-frags (h == 0)
#pragma unroll
    for (int e = 0; e < 16; ++e) { a1[e] = (__bf16)0.0f; a2[e] = (__bf16)0.0f; }

    const float* xg = xg0 + (size_t)btile * gru::T * gru::MT * gru::G;

#pragma unroll 1
    for (int t = 0; t < gru::T; ++t, xg += gru::MT * gru::G) {
        // ---- layer 0: gh = h1 @ w_hh0^T + b_hh0 (K=32 in one WMMA) ----
        v8f g[6];
#pragma unroll
        for (int nt = 0; nt < 6; ++nt)
            g[nt] = __builtin_amdgcn_wmma_f32_16x16x32_bf16(
                false, a1, false, Whh0[nt], (short)0, splat8(bh0[nt]),
                false, false);
        // r,z pre-activations: add streamed xg0 and apply sigmoid
#pragma unroll
        for (int nt = 0; nt < 4; ++nt)
#pragma unroll
            for (int j = 0; j < 8; ++j)
                g[nt][j] = sigmoidf_(
                    g[nt][j] + xg[(j + 8 * hi) * gru::G + nt * 16 + lanelo]);
        // n gate: n = tanh(xn + r * (h@w_hh_n + b_hh_n)); h1 update
#pragma unroll
        for (int c = 0; c < 2; ++c)
#pragma unroll
            for (int j = 0; j < 8; ++j) {
                const float xn =
                    xg[(j + 8 * hi) * gru::G + (4 + c) * 16 + lanelo];
                const float nf = tanhf(xn + g[c][j] * g[4 + c][j]);
                const float z  = g[2 + c][j];
                h1[c][j] = (1.0f - z) * nf + z * h1[c][j];
            }
        a1 = build_a_frag(h1, sA, lanelo, hi);

        // ---- layer 1: xg1 = h1 @ w_ih1^T + b_ih1 ; gh2 = h2 @ w_hh1^T + b_hh1
        v8f p[6];
#pragma unroll
        for (int nt = 0; nt < 6; ++nt)
            p[nt] = __builtin_amdgcn_wmma_f32_16x16x32_bf16(
                false, a1, false, Wih1[nt], (short)0, splat8(bi1[nt]),
                false, false);
#pragma unroll
        for (int nt = 0; nt < 4; ++nt) {
            v8f q = __builtin_amdgcn_wmma_f32_16x16x32_bf16(
                false, a2, false, Whh1[nt], (short)0, splat8(bh1[nt]),
                false, false);
#pragma unroll
            for (int j = 0; j < 8; ++j)
                p[nt][j] = sigmoidf_(p[nt][j] + q[j]);   // r2 / z2
        }
#pragma unroll
        for (int c = 0; c < 2; ++c) {
            v8f qn = __builtin_amdgcn_wmma_f32_16x16x32_bf16(
                false, a2, false, Whh1[4 + c], (short)0, splat8(bh1[4 + c]),
                false, false);
#pragma unroll
            for (int j = 0; j < 8; ++j) {
                const float nf = tanhf(p[4 + c][j] + p[c][j] * qn[j]);
                const float z  = p[2 + c][j];
                h2[c][j] = (1.0f - z) * nf + z * h2[c][j];
            }
        }
        a2 = build_a_frag(h2, sA, lanelo, hi);
    }

    // Final hidden state of layer 1 -> [B][H]
#pragma unroll
    for (int c = 0; c < 2; ++c)
#pragma unroll
        for (int j = 0; j < 8; ++j)
            h2_last[(size_t)(btile * gru::MT + j + 8 * hi) * gru::H +
                    c * 16 + lanelo] = h2[c][j];
}

// ---------------------------------------------------------------------------
// Kernel 3: out[b] = dot(h2_last[b,:], fc_w) + fc_b
// ---------------------------------------------------------------------------
__global__ __launch_bounds__(256) void gru_fc(
    const float* __restrict__ h2_last, const float* __restrict__ fc_w,
    const float* __restrict__ fc_b, float* __restrict__ out)
{
    const int b = blockIdx.x * blockDim.x + threadIdx.x;
    if (b >= gru::B) return;
    const float* h = h2_last + (size_t)b * gru::H;
    float acc = fc_b[0];
#pragma unroll
    for (int i = 0; i < gru::H; ++i) acc = fmaf(h[i], fc_w[i], acc);
    out[b] = acc;
}

// ---------------------------------------------------------------------------
// Launch. Inputs (setup_inputs order):
//  0 x, 1 w_ih_l0, 2 w_hh_l0, 3 b_ih_l0, 4 b_hh_l0,
//  5 w_ih_l1, 6 w_hh_l1, 7 b_ih_l1, 8 b_hh_l1, 9 fc_w, 10 fc_b
// Workspace: xg0 (188,743,680 B) then h2_last (1,048,576 B).
// ---------------------------------------------------------------------------
extern "C" void kernel_launch(void* const* d_in, const int* in_sizes, int n_in,
                              void* d_out, int out_size, void* d_ws,
                              size_t ws_size, hipStream_t stream)
{
    const float* x      = (const float*)d_in[0];
    const float* w_ih0  = (const float*)d_in[1];
    const float* w_hh0  = (const float*)d_in[2];
    const float* b_ih0  = (const float*)d_in[3];
    const float* b_hh0  = (const float*)d_in[4];
    const float* w_ih1  = (const float*)d_in[5];
    const float* w_hh1  = (const float*)d_in[6];
    const float* b_ih1  = (const float*)d_in[7];
    const float* b_hh1  = (const float*)d_in[8];
    const float* fc_w   = (const float*)d_in[9];
    const float* fc_b   = (const float*)d_in[10];
    float*       out    = (float*)d_out;

    float* xg0     = (float*)d_ws;                                  // 189 MB
    float* h2_last = (float*)((char*)d_ws +
                     (size_t)gru::B * gru::T * gru::G * sizeof(float));

    // 30720 (btile,t) tiles, 8 waves per 256-thread block -> 3840 blocks
    gru_xgemm0<<<(gru::B / gru::MT) * gru::T / 8, 256, 0, stream>>>(
        x, w_ih0, b_ih0, xg0);

    // 512 independent scan waves (one per 16 batch rows)
    gru_scan<<<gru::B / gru::MT, 32, 0, stream>>>(
        xg0, w_hh0, b_hh0, w_ih1, b_ih1, w_hh1, b_hh1, h2_last);

    gru_fc<<<gru::B / 256, 256, 0, stream>>>(h2_last, fc_w, fc_b, out);
}
